// Attention_43868795961547
// MI455X (gfx1250) — matrix operand order repaired and checked by reading, not compiled, not run on hardware
//
#include <hip/hip_runtime.h>

typedef __attribute__((ext_vector_type(16))) _Float16 v16h;
typedef __attribute__((ext_vector_type(8)))  _Float16 v8h;
typedef __attribute__((ext_vector_type(8)))  float    v8f;
typedef __attribute__((ext_vector_type(4)))  int      v4i;

#define B_    8
#define NQ    1024
#define NKV   1025
#define NKP   1056     // padded kv length (multiple of 32)
#define H_    8
#define DH    64
#define DIM_  512
#define QKV3  1536
// dim_head^-0.5 * log2(e): folded into Q so softmax uses exp2 directly
#define QSCALE 0.18033688011112042f

union V16H { v16h v; v8h h[2]; };

// ---- optional CDNA5 async global->LDS staging (ASYNCcnt-tracked) ----
#if defined(__has_builtin)
#  if __has_builtin(__builtin_amdgcn_global_load_async_to_lds_b128) && \
      __has_builtin(__builtin_amdgcn_s_wait_asynccnt)
#    define HAVE_ASYNC_LDS 1
#  endif
#endif
#ifndef HAVE_ASYNC_LDS
#  define HAVE_ASYNC_LDS 0
#endif

typedef __attribute__((address_space(1))) v4i* gp_v4i;
typedef __attribute__((address_space(3))) v4i* lp_v4i;

__device__ __forceinline__ void stage16(const _Float16* g, _Float16* l) {
#if HAVE_ASYNC_LDS
  __builtin_amdgcn_global_load_async_to_lds_b128((gp_v4i)(void*)g, (lp_v4i)(void*)l, 0, 0);
#else
  *(v8h*)l = *(const v8h*)g;
#endif
}
__device__ __forceinline__ void stage_wait() {
#if HAVE_ASYNC_LDS
  __builtin_amdgcn_s_wait_asynccnt(0);
#endif
}

// ---- WMMA fragment loaders (gfx1250 wave32 layouts, CDNA5 ISA) ----
// A (16x32 f16): lane l holds row (l&15); halves 0..15 -> K=kb..kb+15,
// where kb = k0 + 16*(l>=16).
__device__ __forceinline__ v16h load_a16(const _Float16* p, int stride, int m0, int k0) {
  int lane = threadIdx.x & 31;
  int row  = m0 + (lane & 15);
  int kb   = k0 + (lane & 16);
  V16H r;
  r.h[0] = *(const v8h*)(p + (size_t)row * stride + kb);
  r.h[1] = *(const v8h*)(p + (size_t)row * stride + kb + 8);
  return r.v;
}
// B (32x16 f16) from [N][K] storage: lane l holds column n0+(l&15),
// contiguous 16 halfs K = k0 + 16*(l>=16) .. +15.
__device__ __forceinline__ v16h load_b16(const _Float16* p, int stride, int n0, int k0) {
  int lane = threadIdx.x & 31;
  int col  = n0 + (lane & 15);
  int kb   = k0 + (lane & 16);
  V16H r;
  r.h[0] = *(const v8h*)(p + (size_t)col * stride + kb);
  r.h[1] = *(const v8h*)(p + (size_t)col * stride + kb + 8);
  return r.v;
}

__device__ __forceinline__ v8f wmma_f16(v16h a, v16h b, v8f c) {
  return __builtin_amdgcn_wmma_f32_16x16x32_f16(false, a, false, b, (short)0, c, false, false);
}

// ---- weight convert + transpose: w[K][N] f32 -> wt[N][K] f16 ----
__global__ void convert_wt_kernel(const float* __restrict__ w, _Float16* __restrict__ wt,
                                  int K, int N) {
  int idx = blockIdx.x * blockDim.x + threadIdx.x;
  if (idx >= K * N) return;
  int k = idx / N, n = idx - k * N;
  wt[(size_t)n * K + k] = (_Float16)w[idx];
}

// ---- layernorm: one wave per 512-wide row, img rows then tab rows ----
__global__ __launch_bounds__(128)
void ln_kernel(const float* __restrict__ img, const float* __restrict__ tab,
               const float* __restrict__ ln_w, const float* __restrict__ ln_b,
               _Float16* __restrict__ xh, _Float16* __restrict__ tabh,
               int nrows_img, int nrows_total) {
  int wave = (blockIdx.x * blockDim.x + threadIdx.x) >> 5;
  int lane = threadIdx.x & 31;
  if (wave >= nrows_total) return;
  const float* src; _Float16* dst;
  if (wave < nrows_img) { src = img + (size_t)wave * DIM_; dst = xh + (size_t)wave * DIM_; }
  else { src = tab + (size_t)(wave - nrows_img) * DIM_; dst = tabh + (size_t)(wave - nrows_img) * DIM_; }
  float v[16]; float s = 0.f, ss = 0.f;
#pragma unroll
  for (int i = 0; i < 16; i++) { v[i] = src[lane + i * 32]; s += v[i]; ss += v[i] * v[i]; }
#pragma unroll
  for (int m = 16; m >= 1; m >>= 1) { s += __shfl_xor(s, m); ss += __shfl_xor(ss, m); }
  float mu  = s * (1.f / DIM_);
  float var = ss * (1.f / DIM_) - mu * mu;
  float rs  = rsqrtf(var + 1e-5f);
#pragma unroll
  for (int i = 0; i < 16; i++) {
    int c = lane + i * 32;
    dst[c] = (_Float16)((v[i] - mu) * rs * ln_w[c] + ln_b[c]);
  }
}

#define LDS_STRIDE 40   // 32-half K-chunk + pad (80B rows, 16B aligned)

// ---- QKV GEMM: X[8192][512] @ Wt[1536][512] -> q / k / vT (f16) ----
// 128M x 64N block, 4 waves, each wave 32M x 64N (acc[2][4], B-frag reuse x2)
__global__ __launch_bounds__(128)
void qkv_gemm_kernel(const _Float16* __restrict__ X, const _Float16* __restrict__ Wt,
                     _Float16* __restrict__ q, _Float16* __restrict__ k,
                     _Float16* __restrict__ vt) {
  __shared__ _Float16 lA[128 * LDS_STRIDE];
  __shared__ _Float16 lB[64 * LDS_STRIDE];
  int tid = threadIdx.x, wave = tid >> 5;
  int m0 = blockIdx.x * 128, n0 = blockIdx.y * 64;
  v8f acc[2][4] = {};
  for (int k0 = 0; k0 < DIM_; k0 += 32) {
#pragma unroll 4
    for (int i = tid; i < 512; i += 128) {
      int row = i >> 2, seg = i & 3;
      stage16(X + (size_t)(m0 + row) * DIM_ + k0 + seg * 8, &lA[row * LDS_STRIDE + seg * 8]);
    }
#pragma unroll 2
    for (int i = tid; i < 256; i += 128) {
      int row = i >> 2, seg = i & 3;
      stage16(Wt + (size_t)(n0 + row) * DIM_ + k0 + seg * 8, &lB[row * LDS_STRIDE + seg * 8]);
    }
    stage_wait();
    __syncthreads();
    v16h a0 = load_a16(lA, LDS_STRIDE, wave * 32, 0);
    v16h a1 = load_a16(lA, LDS_STRIDE, wave * 32 + 16, 0);
#pragma unroll
    for (int j = 0; j < 4; j++) {
      v16h b = load_b16(lB, LDS_STRIDE, j * 16, 0);
      acc[0][j] = wmma_f16(a0, b, acc[0][j]);
      acc[1][j] = wmma_f16(a1, b, acc[1][j]);
    }
    __syncthreads();
  }
  int lane = tid & 31;
  int rowoff = (lane & 16) >> 1, cbase = lane & 15;
#pragma unroll
  for (int mi = 0; mi < 2; mi++) {
#pragma unroll
    for (int j = 0; j < 4; j++) {
      int jc = n0 + j * 16 + cbase;
#pragma unroll
      for (int r = 0; r < 8; r++) {
        int gm = m0 + wave * 32 + mi * 16 + r + rowoff;
        int b  = gm >> 10, n = gm & 1023;
        float v = acc[mi][j][r];
        if (jc < 512) {                       // Q (prescaled by scale*log2e)
          int head = jc >> 6, d = jc & 63;
          q[(((size_t)(b * H_ + head)) * NQ + n) * DH + d] = (_Float16)(v * QSCALE);
        } else if (jc < 1024) {               // K: [b,h,nkv,64]
          int jj = jc - 512; int head = jj >> 6, d = jj & 63;
          k[(((size_t)(b * H_ + head)) * NKP + n) * DH + d] = (_Float16)v;
        } else {                              // V transposed: [b,h,64,nkv]
          int jj = jc - 1024; int head = jj >> 6, d = jj & 63;
          vt[(((size_t)(b * H_ + head)) * DH + d) * NKP + n] = (_Float16)v;
        }
      }
    }
  }
}

// ---- tab-row QKV (8 rows) appended at kv index 1024 ----
__global__ __launch_bounds__(256)
void tab_qkv_kernel(const _Float16* __restrict__ tabh, const float* __restrict__ w,
                    _Float16* __restrict__ k, _Float16* __restrict__ vt) {
  __shared__ float t[DIM_];
  int b = blockIdx.x, tid = threadIdx.x;
  for (int i = tid; i < DIM_; i += 256) t[i] = (float)tabh[(size_t)b * DIM_ + i];
  __syncthreads();
  for (int j = 512 + tid; j < QKV3; j += 256) {
    float acc = 0.f;
    for (int kk = 0; kk < DIM_; kk++) acc += t[kk] * w[(size_t)kk * QKV3 + j];
    if (j < 1024) {
      int jj = j - 512, head = jj >> 6, d = jj & 63;
      k[(((size_t)(b * H_ + head)) * NKP + NQ) * DH + d] = (_Float16)acc;
    } else {
      int jj = j - 1024, head = jj >> 6, d = jj & 63;
      vt[(((size_t)(b * H_ + head)) * DH + d) * NKP + NQ] = (_Float16)acc;
    }
  }
}

// ---- flash attention: WG = 64 q rows (4 waves x 16), key blocks of 32 ----
#define KSTR 72   // K tile [32 kv][64 d + pad]   (144B rows)
#define VSTR 40   // vT tile [64 d][32 kv + pad]  (80B rows)
#define PSTR 40   // per-wave P patch [16][32 + pad]
__global__ __launch_bounds__(128)
void attn_kernel(const _Float16* __restrict__ q, const _Float16* __restrict__ kk,
                 const _Float16* __restrict__ vt, _Float16* __restrict__ attn_out) {
  __shared__ _Float16 lK[32 * KSTR];
  __shared__ _Float16 lV[64 * VSTR];
  __shared__ _Float16 lP[4 * 16 * PSTR];
  int tid = threadIdx.x, wave = tid >> 5, lane = tid & 31;
  int bh = blockIdx.y;
  int q0 = blockIdx.x * 64 + wave * 16;
  const _Float16* qbase = q  + (size_t)bh * NQ * DH;
  const _Float16* kbase = kk + (size_t)bh * NKP * DH;
  const _Float16* vbase = vt + (size_t)bh * DH * NKP;
  // Q fragments held in registers for the whole kv sweep (already scaled)
  v16h aq0 = load_a16(qbase, DH, q0, 0);
  v16h aq1 = load_a16(qbase, DH, q0, 32);
  v8f acc[4] = {};
  float mrow[8], lrow[8];
#pragma unroll
  for (int r = 0; r < 8; r++) { mrow[r] = -1e30f; lrow[r] = 0.f; }
  _Float16* myP = lP + wave * 16 * PSTR;
  int cbase = lane & 15, rowoff = (lane & 16) >> 1;

  for (int kb = 0; kb < NKV; kb += 32) {
    // stage K rows kb..kb+31 (global rows contiguous) and vT cols kb..kb+31
#pragma unroll 2
    for (int i = tid; i < 256; i += 128) {
      int kr = i >> 3, ks = i & 7;
      stage16(kbase + (size_t)(kb + kr) * DH + ks * 8, &lK[kr * KSTR + ks * 8]);
      int vr = i >> 2, vs = i & 3;   // i<256 covers 64 rows x 4 segs
      stage16(vbase + (size_t)vr * NKP + kb + vs * 8, &lV[vr * VSTR + vs * 8]);
    }
    stage_wait();
    __syncthreads();
    // S = Q K^T  (two 16-col tiles, K-dim 64 = 2 wmma each)
    v8f s0 = {}, s1 = {};
    s0 = wmma_f16(aq0, load_b16(lK, KSTR, 0, 0),  s0);
    s0 = wmma_f16(aq1, load_b16(lK, KSTR, 0, 32), s0);
    s1 = wmma_f16(aq0, load_b16(lK, KSTR, 16, 0),  s1);
    s1 = wmma_f16(aq1, load_b16(lK, KSTR, 16, 32), s1);
    if (kb + 32 > NKV) {  // tail mask (kv pad)
#pragma unroll
      for (int r = 0; r < 8; r++) {
        if (kb + cbase      >= NKV) s0[r] = -1e30f;
        if (kb + 16 + cbase >= NKV) s1[r] = -1e30f;
      }
    }
    // online softmax (base-2 domain; row lives on 16 lanes of one C vgpr)
    float alpha[8];
#pragma unroll
    for (int r = 0; r < 8; r++) {
      float mb = fmaxf(s0[r], s1[r]);
      mb = fmaxf(mb, __shfl_xor(mb, 1)); mb = fmaxf(mb, __shfl_xor(mb, 2));
      mb = fmaxf(mb, __shfl_xor(mb, 4)); mb = fmaxf(mb, __shfl_xor(mb, 8));
      float mn = fmaxf(mrow[r], mb);
      alpha[r] = exp2f(mrow[r] - mn);
      mrow[r] = mn;
      float p0 = exp2f(s0[r] - mn), p1 = exp2f(s1[r] - mn);
      s0[r] = p0; s1[r] = p1;
      float rs = p0 + p1;
      rs += __shfl_xor(rs, 1); rs += __shfl_xor(rs, 2);
      rs += __shfl_xor(rs, 4); rs += __shfl_xor(rs, 8);
      lrow[r] = lrow[r] * alpha[r] + rs;
    }
    // C-layout P -> LDS -> A-layout fragment (per-wave patch, no barrier)
#pragma unroll
    for (int r = 0; r < 8; r++) {
      myP[(r + rowoff) * PSTR + cbase]      = (_Float16)s0[r];
      myP[(r + rowoff) * PSTR + 16 + cbase] = (_Float16)s1[r];
    }
    v16h pa = load_a16(myP, PSTR, 0, 0);
#pragma unroll
    for (int j = 0; j < 4; j++) {
#pragma unroll
      for (int r = 0; r < 8; r++) acc[j][r] *= alpha[r];
      acc[j] = wmma_f16(pa, load_b16(lV, VSTR, j * 16, 0), acc[j]);
    }
    __syncthreads();
  }
  // normalize + store to [b, n, h*64+d] f16 (feeds output GEMM)
  int b = bh >> 3, h = bh & 7;
#pragma unroll
  for (int r = 0; r < 8; r++) {
    float inv = 1.0f / lrow[r];
    int gm = q0 + r + rowoff;
#pragma unroll
    for (int j = 0; j < 4; j++) {
      int col = h * DH + j * 16 + cbase;
      attn_out[((size_t)b * NQ + gm) * DIM_ + col] = (_Float16)(acc[j][r] * inv);
    }
  }
}

// ---- output GEMM: attn[8192][512] @ WoT[512][512] + bias -> f32 out ----
__global__ __launch_bounds__(128)
void out_gemm_kernel(const _Float16* __restrict__ A, const _Float16* __restrict__ Wt,
                     const float* __restrict__ bias, float* __restrict__ out) {
  __shared__ _Float16 lA[128 * LDS_STRIDE];
  __shared__ _Float16 lB[64 * LDS_STRIDE];
  int tid = threadIdx.x, wave = tid >> 5;
  int m0 = blockIdx.x * 128, n0 = blockIdx.y * 64;
  v8f acc[2][4] = {};
  for (int k0 = 0; k0 < DIM_; k0 += 32) {
#pragma unroll 4
    for (int i = tid; i < 512; i += 128) {
      int row = i >> 2, seg = i & 3;
      stage16(A + (size_t)(m0 + row) * DIM_ + k0 + seg * 8, &lA[row * LDS_STRIDE + seg * 8]);
    }
#pragma unroll 2
    for (int i = tid; i < 256; i += 128) {
      int row = i >> 2, seg = i & 3;
      stage16(Wt + (size_t)(n0 + row) * DIM_ + k0 + seg * 8, &lB[row * LDS_STRIDE + seg * 8]);
    }
    stage_wait();
    __syncthreads();
    v16h a0 = load_a16(lA, LDS_STRIDE, wave * 32, 0);
    v16h a1 = load_a16(lA, LDS_STRIDE, wave * 32 + 16, 0);
#pragma unroll
    for (int j = 0; j < 4; j++) {
      v16h b = load_b16(lB, LDS_STRIDE, j * 16, 0);
      acc[0][j] = wmma_f16(a0, b, acc[0][j]);
      acc[1][j] = wmma_f16(a1, b, acc[1][j]);
    }
    __syncthreads();
  }
  int lane = tid & 31;
  int rowoff = (lane & 16) >> 1, cbase = lane & 15;
#pragma unroll
  for (int mi = 0; mi < 2; mi++) {
#pragma unroll
    for (int j = 0; j < 4; j++) {
      int jc = n0 + j * 16 + cbase;
      float bj = bias[jc];
#pragma unroll
      for (int r = 0; r < 8; r++) {
        int gm = m0 + wave * 32 + mi * 16 + r + rowoff;
        out[(size_t)gm * DIM_ + jc] = acc[mi][j][r] + bj;
      }
    }
  }
}

extern "C" void kernel_launch(void* const* d_in, const int* in_sizes, int n_in,
                              void* d_out, int out_size, void* d_ws, size_t ws_size,
                              hipStream_t stream) {
  (void)in_sizes; (void)n_in; (void)out_size; (void)ws_size;
  const float* img      = (const float*)d_in[0];
  const float* tab      = (const float*)d_in[1];
  const float* w_qkv    = (const float*)d_in[2];
  const float* w_tabqkv = (const float*)d_in[3];
  const float* w_out    = (const float*)d_in[4];
  const float* b_out    = (const float*)d_in[5];
  const float* ln_w     = (const float*)d_in[6];
  const float* ln_b     = (const float*)d_in[7];
  float* out = (float*)d_out;

  char* ws = (char*)d_ws;
  size_t off = 0;
  auto take = [&](size_t bytes) { char* p = ws + off; off = (off + bytes + 255) & ~(size_t)255; return p; };
  _Float16* xh    = (_Float16*)take((size_t)B_ * NQ * DIM_ * 2);     // 8 MB
  _Float16* tabh  = (_Float16*)take((size_t)B_ * DIM_ * 2);
  _Float16* wqkvt = (_Float16*)take((size_t)QKV3 * DIM_ * 2);        // 1.5 MB
  _Float16* wot   = (_Float16*)take((size_t)DIM_ * DIM_ * 2);        // 0.5 MB
  _Float16* qh    = (_Float16*)take((size_t)B_ * H_ * NQ * DH * 2);  // 8 MB
  _Float16* kh    = (_Float16*)take((size_t)B_ * H_ * NKP * DH * 2); // 8.25 MB
  _Float16* vth   = (_Float16*)take((size_t)B_ * H_ * DH * NKP * 2); // 8.25 MB
  _Float16* ath   = (_Float16*)take((size_t)B_ * NQ * DIM_ * 2);     // 8 MB

  convert_wt_kernel<<<(DIM_ * QKV3 + 255) / 256, 256, 0, stream>>>(w_qkv, wqkvt, DIM_, QKV3);
  convert_wt_kernel<<<(DIM_ * DIM_ + 255) / 256, 256, 0, stream>>>(w_out, wot, DIM_, DIM_);

  int rows_img = B_ * NQ, rows_total = rows_img + B_;
  ln_kernel<<<(rows_total + 3) / 4, 128, 0, stream>>>(img, tab, ln_w, ln_b, xh, tabh,
                                                      rows_img, rows_total);

  qkv_gemm_kernel<<<dim3(rows_img / 128, QKV3 / 64), 128, 0, stream>>>(xh, wqkvt, qh, kh, vth);
  tab_qkv_kernel<<<B_, 256, 0, stream>>>(tabh, w_tabqkv, kh, vth);

  attn_kernel<<<dim3(NQ / 64, B_ * H_), 128, 0, stream>>>(qh, kh, vth, ath);

  out_gemm_kernel<<<dim3(rows_img / 128, DIM_ / 64), 128, 0, stream>>>(ath, wot, b_out, out);
}
